// Phase8Layer_16870631539351
// MI455X (gfx1250) — matrix operand order repaired and checked by reading, not compiled, not run on hardware
//
#include <hip/hip_runtime.h>
#include <cmath>
#include <cstdint>

// ---------------- problem constants ----------------
#define B_    4
#define S_    2048
#define D_    2048
#define H_    16
#define HD_   128
#define RANK_ 256
#define HID_  8192
#define BS_   (B_ * S_)

typedef __bf16 bf16;
typedef __attribute__((ext_vector_type(16))) __bf16 v16bf;
typedef __attribute__((ext_vector_type(8)))  __bf16 v8bf;
typedef __attribute__((ext_vector_type(4)))  __bf16 v4bf;
typedef __attribute__((ext_vector_type(8)))  float   v8f;
typedef __attribute__((ext_vector_type(4)))  unsigned int u32x4;
typedef __attribute__((ext_vector_type(8)))  int i32x8;
typedef __attribute__((ext_vector_type(4)))  int i32x4;

// ---------------- GEMM tiling ----------------
#define BM 128
#define BN 128
#define BK 32
#define LDA_S 40     // BK + 8 pad (bf16) -> 80B row stride  (TDM: 16DW + 4DW pad)
#define KVLD  136    // HD + 8 pad       -> 272B row stride  (TDM: 64DW + 4DW pad)

enum { EPI_NONE = 0, EPI_ELU1 = 1, EPI_GELU = 2, EPI_RESID = 3 };

// low 32 bits of a flat shared-memory pointer == LDS byte offset (ISA 10.2 aperture rule)
__device__ __forceinline__ unsigned lds_off(const void* p) {
  return (unsigned)(unsigned long long)(uintptr_t)p;
}

// ---------------- Tensor Data Mover: 2D tile global->LDS with LDS row padding ----------------
// D# group0: count=1, lds_addr, 57b global addr, type=2. group1: data_size=2B,
// pad_enable, pad_interval/pad_amount codes, tensor dims, tile dims, dim0 stride.
__device__ __forceinline__ void tdm_load_2d(unsigned ldsoff, const void* gptr,
                                            unsigned tile_w, unsigned tile_h,
                                            unsigned long long stride_elems,
                                            unsigned pad_int, unsigned pad_amt) {
  unsigned long long ga = (unsigned long long)(uintptr_t)gptr;
  u32x4 g0;
  g0[0] = 1u;                                                   // count=1, user D#
  g0[1] = ldsoff;                                               // lds_addr [63:32]
  g0[2] = (unsigned)ga;                                         // global_addr [95:64]
  g0[3] = (unsigned)((ga >> 32) & 0x01ffffffULL) | (2u << 30);  // [120:96], type=2
  i32x8 g1;
  g1[0] = (int)((1u << 16) | (1u << 20) | (pad_int << 22) | (pad_amt << 25));
  unsigned td0 = (unsigned)stride_elems;  // tensor_dim0 (row length)
  unsigned td1 = tile_h;                  // tensor_dim1
  g1[1] = (int)((td0 & 0xffffu) << 16);                      // [63:48]=td0.lo16
  g1[2] = (int)((td0 >> 16) | ((td1 & 0xffffu) << 16));      // [79:64],[95:80]
  g1[3] = (int)((td1 >> 16) | (tile_w << 16));               // [111:96],[127:112]=tile_dim0
  g1[4] = (int)(tile_h & 0xffffu);                           // [143:128]=tile_dim1, tile_dim2=0
  g1[5] = (int)(unsigned)(stride_elems & 0xffffffffULL);     // dim0_stride lo
  g1[6] = (int)(unsigned)((stride_elems >> 32) & 0xffffULL); // dim0_stride hi
  g1[7] = 0;
  i32x4 g2 = {0, 0, 0, 0};
  i32x4 g3 = {0, 0, 0, 0};
#if defined(__clang_major__) && (__clang_major__ >= 23)
  i32x8 g4 = {0, 0, 0, 0, 0, 0, 0, 0};
  __builtin_amdgcn_tensor_load_to_lds(g0, g1, g2, g3, g4, 0);
#else
  __builtin_amdgcn_tensor_load_to_lds(g0, g1, g2, g3, 0);
#endif
}

// A fragment (16x32 bf16) from row-major LDS tile, row stride ldk.
// lanes 0-15 row M=lane: K=[0..7],[16..23]; lanes 16-31 row M=lane-16: K=[8..15],[24..31].
__device__ __forceinline__ v16bf load_afrag(const bf16* base, int lrow, int lhi, int ldk) {
  const bf16* p = base + lrow * ldk + lhi * 8;
  v8bf lo = *(const v8bf*)p;
  v8bf hi = *(const v8bf*)(p + 16);
  v16bf a;
#pragma unroll
  for (int j = 0; j < 8; ++j) { a[j] = lo[j]; a[j + 8] = hi[j]; }
  return a;
}

// B fragment (32x16) from transposed LDS tile BsT[n][k], row stride ldk.
// lanes 0-15 col N=lane: K=[0..7],[16..23]; lanes 16-31: K=[8..15],[24..31].
__device__ __forceinline__ v16bf load_bfrag(const bf16* base, int lrow, int lhi, int ldk) {
  const bf16* p = base + lrow * ldk + lhi * 8;
  v8bf lo = *(const v8bf*)p;
  v8bf hi = *(const v8bf*)(p + 16);
  v16bf b;
#pragma unroll
  for (int j = 0; j < 8; ++j) { b[j] = lo[j]; b[j + 8] = hi[j]; }
  return b;
}

// ---------------- generic bf16 GEMM: C = epi(A[MxK] @ B[KxN] + bias) (+resid) ----------------
// TDM-fed double-buffered A tiles; transposed B staged with packed b64 LDS stores.
template <int EPI>
__global__ __launch_bounds__(256) void gemm_bf16_k(
    const bf16* __restrict__ A, const bf16* __restrict__ Bm,
    const float* __restrict__ bias, const float* __restrict__ resid,
    float* __restrict__ outF, bf16* __restrict__ outH, int M, int N, int K) {
  __shared__ __align__(16) bf16 As[2][BM * LDA_S];
  __shared__ __align__(16) bf16 BsT[2][BN * LDA_S];
  const int t = threadIdx.x;
  const int lane = t & 31, w = t >> 5;
  const int wm = (w & 3) * 32, wn = (w >> 2) * 64;
  const int lrow = lane & 15, lhi = lane >> 4;
  const int m0 = blockIdx.y * BM, n0 = blockIdx.x * BN;

  // transpose-stage one 32x128 B tile: 4 k-rows x 4 cols per thread, b64 LDS stores
  auto stage_b = [&](int buf, int kk) {
    const int kq = (t >> 5) * 4;   // 0..28
    const int n4 = (t & 31) * 4;   // 0..124
    const bf16* src = Bm + (size_t)(kk + kq) * N + n0 + n4;
    v4bf r0 = *(const v4bf*)(src);
    v4bf r1 = *(const v4bf*)(src + (size_t)N);
    v4bf r2 = *(const v4bf*)(src + 2 * (size_t)N);
    v4bf r3 = *(const v4bf*)(src + 3 * (size_t)N);
#pragma unroll
    for (int j = 0; j < 4; ++j) {
      v4bf p; p[0] = r0[j]; p[1] = r1[j]; p[2] = r2[j]; p[3] = r3[j];
      *(v4bf*)&BsT[buf][(n4 + j) * LDA_S + kq] = p;
    }
  };

  // prologue: TDM A tile 0 (wave 0), stage B tile 0 (all)
  if (w == 0)
    tdm_load_2d(lds_off(&As[0][0]), A + (size_t)m0 * K, BK, BM,
                (unsigned long long)K, 3, 3);  // 16DW rows + 4DW pad -> LDA_S
  stage_b(0, 0);
  if (w == 0) __builtin_amdgcn_s_wait_tensorcnt((short)0);
  __syncthreads();

  v8f acc[2][4] = {};
  int cur = 0;
  for (int k0 = 0; k0 < K; k0 += BK) {
    const int nxt = cur ^ 1;
    if (k0 + BK < K) {  // prefetch next tiles while computing this one
      if (w == 0)
        tdm_load_2d(lds_off(&As[nxt][0]), A + (size_t)m0 * K + (k0 + BK), BK, BM,
                    (unsigned long long)K, 3, 3);
      stage_b(nxt, k0 + BK);
    }
    v16bf af[2], bfv[4];
#pragma unroll
    for (int mi = 0; mi < 2; ++mi)
      af[mi] = load_afrag(&As[cur][(wm + mi * 16) * LDA_S], lrow, lhi, LDA_S);
#pragma unroll
    for (int ni = 0; ni < 4; ++ni)
      bfv[ni] = load_bfrag(&BsT[cur][(wn + ni * 16) * LDA_S], lrow, lhi, LDA_S);
#pragma unroll
    for (int mi = 0; mi < 2; ++mi)
#pragma unroll
      for (int ni = 0; ni < 4; ++ni)
        acc[mi][ni] = __builtin_amdgcn_wmma_f32_16x16x32_bf16(
            false, af[mi], false, bfv[ni], (short)0, acc[mi][ni], false, false);
    if (w == 0) __builtin_amdgcn_s_wait_tensorcnt((short)0);
    __syncthreads();
    cur = nxt;
  }
  // epilogue: C layout — lanes 0-15 row=r, lanes 16-31 row=8+r; col=lane&15
#pragma unroll
  for (int mi = 0; mi < 2; ++mi)
#pragma unroll
    for (int ni = 0; ni < 4; ++ni) {
      const int col = n0 + wn + ni * 16 + lrow;
      const float bv = bias[col];
#pragma unroll
      for (int r = 0; r < 8; ++r) {
        const int row = m0 + wm + mi * 16 + 8 * lhi + r;
        float v = acc[mi][ni][r] + bv;
        if (EPI == EPI_ELU1) {
          v = (v > 0.f) ? (v + 1.f) : __expf(v);  // elu(x)+1
        } else if (EPI == EPI_GELU) {
          float u = 0.7978845608028654f * (v + 0.044715f * v * v * v);
          v = 0.5f * v * (1.f + tanhf(u));
        }
        if (EPI == EPI_RESID)
          outF[(size_t)row * N + col] = v + resid[(size_t)row * N + col];
        else
          outH[(size_t)row * N + col] = (bf16)v;
      }
    }
}

// ---------------- kv = K^T V per (b,h): 128x128, reduce over S (double-buffered) ----------------
__global__ __launch_bounds__(256) void kv_kernel(const bf16* __restrict__ Kb,
                                                 const bf16* __restrict__ Vb,
                                                 bf16* __restrict__ kv16) {
  __shared__ __align__(16) bf16 KsT[2][HD_ * LDA_S];
  __shared__ __align__(16) bf16 VsT[2][HD_ * LDA_S];
  const int t = threadIdx.x, lane = t & 31, w = t >> 5;
  const int wm = (w & 3) * 32, wn = (w >> 2) * 64;
  const int lrow = lane & 15, lhi = lane >> 4;
  const int bh = blockIdx.x, b = bh / H_, h = bh % H_;

  auto stage = [&](int buf, int s0) {
    const int sq = (t >> 5) * 4;   // 0..28
    const int d4 = (t & 31) * 4;   // 0..124
    const size_t base = (size_t)(b * S_ + s0 + sq) * D_ + h * HD_ + d4;
    v4bf k0v = *(const v4bf*)(Kb + base);
    v4bf k1v = *(const v4bf*)(Kb + base + D_);
    v4bf k2v = *(const v4bf*)(Kb + base + 2 * D_);
    v4bf k3v = *(const v4bf*)(Kb + base + 3 * D_);
    v4bf v0v = *(const v4bf*)(Vb + base);
    v4bf v1v = *(const v4bf*)(Vb + base + D_);
    v4bf v2v = *(const v4bf*)(Vb + base + 2 * D_);
    v4bf v3v = *(const v4bf*)(Vb + base + 3 * D_);
#pragma unroll
    for (int j = 0; j < 4; ++j) {
      v4bf pk; pk[0] = k0v[j]; pk[1] = k1v[j]; pk[2] = k2v[j]; pk[3] = k3v[j];
      *(v4bf*)&KsT[buf][(d4 + j) * LDA_S + sq] = pk;
      v4bf pv; pv[0] = v0v[j]; pv[1] = v1v[j]; pv[2] = v2v[j]; pv[3] = v3v[j];
      *(v4bf*)&VsT[buf][(d4 + j) * LDA_S + sq] = pv;
    }
  };

  stage(0, 0);
  __syncthreads();
  v8f acc[2][4] = {};
  int cur = 0;
  for (int s0 = 0; s0 < S_; s0 += BK) {
    const int nxt = cur ^ 1;
    if (s0 + BK < S_) stage(nxt, s0 + BK);
    v16bf af[2], bfv[4];
#pragma unroll
    for (int mi = 0; mi < 2; ++mi)
      af[mi] = load_afrag(&KsT[cur][(wm + mi * 16) * LDA_S], lrow, lhi, LDA_S);
#pragma unroll
    for (int ni = 0; ni < 4; ++ni)
      bfv[ni] = load_bfrag(&VsT[cur][(wn + ni * 16) * LDA_S], lrow, lhi, LDA_S);
#pragma unroll
    for (int mi = 0; mi < 2; ++mi)
#pragma unroll
      for (int ni = 0; ni < 4; ++ni)
        acc[mi][ni] = __builtin_amdgcn_wmma_f32_16x16x32_bf16(
            false, af[mi], false, bfv[ni], (short)0, acc[mi][ni], false, false);
    __syncthreads();
    cur = nxt;
  }
#pragma unroll
  for (int mi = 0; mi < 2; ++mi)
#pragma unroll
    for (int ni = 0; ni < 4; ++ni) {
      const int col = wn + ni * 16 + lrow;
#pragma unroll
      for (int r = 0; r < 8; ++r) {
        const int row = wm + mi * 16 + 8 * lhi + r;
        kv16[(size_t)bh * HD_ * HD_ + row * HD_ + col] = (bf16)acc[mi][ni][r];
      }
    }
}

// ---------------- ksum[b,h,d] = sum_s K[b,s,h,d] ----------------
__global__ __launch_bounds__(128) void colsum_kernel(const bf16* __restrict__ Kb,
                                                     float* __restrict__ ksum) {
  const int bh = blockIdx.x, b = bh / H_, h = bh % H_, d = threadIdx.x;
  float s = 0.f;
  for (int ss = 0; ss < S_; ++ss)
    s += (float)Kb[(size_t)(b * S_ + ss) * D_ + h * HD_ + d];
  ksum[bh * HD_ + d] = s;
}

// ---------------- att = (Q @ kv) / (Q . ksum + eps) per (b,h,s-tile) ----------------
// Whole 128x128 Q block DMA'd by TDM once; 4 barrier-free WMMA K-steps.
__global__ __launch_bounds__(256) void attn_kernel(const bf16* __restrict__ Qb,
                                                   const bf16* __restrict__ kv16,
                                                   const float* __restrict__ ksum,
                                                   bf16* __restrict__ att) {
  __shared__ __align__(16) bf16 Qs[BM * KVLD];    // Q[s][d], padded rows
  __shared__ __align__(16) bf16 kvT[HD_ * KVLD];  // kvT[e][d]
  __shared__ float ks_s[HD_];
  __shared__ float den_s[BM];
  const int t = threadIdx.x, lane = t & 31, w = t >> 5;
  const int wm = (w & 3) * 32, wn = (w >> 2) * 64;
  const int lrow = lane & 15, lhi = lane >> 4;
  const int bh = blockIdx.x, b = bh / H_, h = bh % H_;
  const int s0 = blockIdx.y * BM;
  const bf16* qbase = Qb + (size_t)(b * S_ + s0) * D_ + h * HD_;

  if (w == 0)
    tdm_load_2d(lds_off(&Qs[0]), qbase, HD_, BM, (unsigned long long)D_,
                5, 3);  // 64DW rows + 4DW pad -> KVLD
  // stage kv transposed (packed b64 stores)
  for (int i = t; i < (HD_ / 4) * (HD_ / 4); i += 256) {
    const int dq = (i >> 5) * 4;   // 0..124
    const int e4 = (i & 31) * 4;   // 0..124
    const bf16* src = kv16 + (size_t)bh * HD_ * HD_ + (size_t)dq * HD_ + e4;
    v4bf r0 = *(const v4bf*)(src);
    v4bf r1 = *(const v4bf*)(src + HD_);
    v4bf r2 = *(const v4bf*)(src + 2 * HD_);
    v4bf r3 = *(const v4bf*)(src + 3 * HD_);
#pragma unroll
    for (int j = 0; j < 4; ++j) {
      v4bf p; p[0] = r0[j]; p[1] = r1[j]; p[2] = r2[j]; p[3] = r3[j];
      *(v4bf*)&kvT[(e4 + j) * KVLD + dq] = p;
    }
  }
  if (t < HD_) ks_s[t] = ksum[bh * HD_ + t];
  __syncthreads();

  // per-row denominator (global Q read, independent of TDM)
  if (t < BM) {
    const bf16* qr = qbase + (size_t)t * D_;
    float d = 0.f;
#pragma unroll 4
    for (int c = 0; c < HD_; c += 8) {
      v8bf qv = *(const v8bf*)(qr + c);
#pragma unroll
      for (int j = 0; j < 8; ++j) d += (float)qv[j] * ks_s[c + j];
    }
    den_s[t] = d + 1e-6f;
  }
  if (w == 0) __builtin_amdgcn_s_wait_tensorcnt((short)0);
  __syncthreads();

  v8f acc[2][4] = {};
#pragma unroll
  for (int kc = 0; kc < HD_; kc += BK) {
    v16bf af[2], bfv[4];
#pragma unroll
    for (int mi = 0; mi < 2; ++mi)
      af[mi] = load_afrag(&Qs[(wm + mi * 16) * KVLD + kc], lrow, lhi, KVLD);
#pragma unroll
    for (int ni = 0; ni < 4; ++ni)
      bfv[ni] = load_bfrag(&kvT[(wn + ni * 16) * KVLD + kc], lrow, lhi, KVLD);
#pragma unroll
    for (int mi = 0; mi < 2; ++mi)
#pragma unroll
      for (int ni = 0; ni < 4; ++ni)
        acc[mi][ni] = __builtin_amdgcn_wmma_f32_16x16x32_bf16(
            false, af[mi], false, bfv[ni], (short)0, acc[mi][ni], false, false);
  }
#pragma unroll
  for (int mi = 0; mi < 2; ++mi)
#pragma unroll
    for (int ni = 0; ni < 4; ++ni) {
      const int col = wn + ni * 16 + lrow;
#pragma unroll
      for (int r = 0; r < 8; ++r) {
        const int rl = wm + mi * 16 + 8 * lhi + r;
        float v = acc[mi][ni][r] / den_s[rl];
        att[(size_t)(b * S_ + s0 + rl) * D_ + h * HD_ + col] = (bf16)v;
      }
    }
}

// ---------------- LayerNorm: fp32 in -> bf16 out ----------------
__global__ __launch_bounds__(256) void ln_kernel(const float* __restrict__ x,
                                                 const float* __restrict__ g,
                                                 const float* __restrict__ bb,
                                                 bf16* __restrict__ out) {
  __shared__ float rs[256], rs2[256];
  const int t = threadIdx.x;
  const size_t base = (size_t)blockIdx.x * D_;
  float s = 0.f, s2 = 0.f;
  for (int i = t; i < D_; i += 256) {
    float v = x[base + i];
    s += v; s2 += v * v;
  }
  rs[t] = s; rs2[t] = s2;
  __syncthreads();
  for (int o = 128; o > 0; o >>= 1) {
    if (t < o) { rs[t] += rs[t + o]; rs2[t] += rs2[t + o]; }
    __syncthreads();
  }
  const float mu = rs[0] * (1.f / D_);
  const float var = rs2[0] * (1.f / D_) - mu * mu;
  const float rstd = rsqrtf(var + 1e-5f);
  for (int i = t; i < D_; i += 256)
    out[base + i] = (bf16)((x[base + i] - mu) * rstd * g[i] + bb[i]);
}

// ---------------- fp32 -> bf16 conversion ----------------
__global__ __launch_bounds__(256) void cvt_kernel(const float* __restrict__ in,
                                                  bf16* __restrict__ out, long n) {
  long i = (long)blockIdx.x * blockDim.x + threadIdx.x;
  long stride = (long)gridDim.x * blockDim.x;
  for (; i < n; i += stride) out[i] = (bf16)in[i];
}

// ---------------- host: full block pipeline ----------------
extern "C" void kernel_launch(void* const* d_in, const int* in_sizes, int n_in,
                              void* d_out, int out_size, void* d_ws, size_t ws_size,
                              hipStream_t stream) {
  (void)in_sizes; (void)n_in; (void)out_size; (void)ws_size;
  const float* x    = (const float*)d_in[0];
  const float* Wq   = (const float*)d_in[1];
  const float* bq   = (const float*)d_in[2];
  const float* Wk   = (const float*)d_in[3];
  const float* bk   = (const float*)d_in[4];
  const float* Wv   = (const float*)d_in[5];
  const float* bv   = (const float*)d_in[6];
  const float* Wo   = (const float*)d_in[7];
  const float* bo   = (const float*)d_in[8];
  const float* ln1g = (const float*)d_in[9];
  const float* ln1b = (const float*)d_in[10];
  const float* ln2g = (const float*)d_in[11];
  const float* ln2b = (const float*)d_in[12];
  const float* W1   = (const float*)d_in[13];
  const float* b1   = (const float*)d_in[14];
  const float* W2   = (const float*)d_in[15];
  const float* b2   = (const float*)d_in[16];
  const float* W3   = (const float*)d_in[17];
  const float* b3   = (const float*)d_in[18];
  const float* W4   = (const float*)d_in[19];
  const float* b4   = (const float*)d_in[20];
  float* out = (float*)d_out;

  char* ws = (char*)d_ws;
  size_t off = 0;
  auto take = [&](size_t bytes) -> char* {
    char* p = ws + off;
    off = (off + bytes + 255) & ~(size_t)255;
    return p;
  };
  bf16* Wq16 = (bf16*)take((size_t)D_ * D_ * 2);
  bf16* Wk16 = (bf16*)take((size_t)D_ * D_ * 2);
  bf16* Wv16 = (bf16*)take((size_t)D_ * D_ * 2);
  bf16* Wo16 = (bf16*)take((size_t)D_ * D_ * 2);
  bf16* W116 = (bf16*)take((size_t)D_ * RANK_ * 2);
  bf16* W216 = (bf16*)take((size_t)RANK_ * HID_ * 2);
  bf16* W316 = (bf16*)take((size_t)HID_ * RANK_ * 2);
  bf16* W416 = (bf16*)take((size_t)RANK_ * D_ * 2);
  bf16* h1   = (bf16*)take((size_t)BS_ * D_ * 2);
  bf16* Qb   = (bf16*)take((size_t)BS_ * D_ * 2);
  bf16* Kb   = (bf16*)take((size_t)BS_ * D_ * 2);
  bf16* Vb   = (bf16*)take((size_t)BS_ * D_ * 2);
  bf16* kv16 = (bf16*)take((size_t)B_ * H_ * HD_ * HD_ * 2);
  float* ksum = (float*)take((size_t)B_ * H_ * HD_ * 4);
  bf16* attb = (bf16*)take((size_t)BS_ * D_ * 2);
  float* x2  = (float*)take((size_t)BS_ * D_ * 4);
  bf16* h2   = (bf16*)take((size_t)BS_ * D_ * 2);
  bf16* f1   = (bf16*)take((size_t)BS_ * RANK_ * 2);
  bf16* f2   = (bf16*)take((size_t)BS_ * HID_ * 2);
  bf16* f3   = (bf16*)take((size_t)BS_ * RANK_ * 2);

  auto cvt = [&](const float* src, bf16* dst, size_t n) {
    int blocks = (int)((n + 256 * 8 - 1) / (256 * 8));
    if (blocks > 4096) blocks = 4096;
    cvt_kernel<<<blocks, 256, 0, stream>>>(src, dst, (long)n);
  };
  cvt(Wq, Wq16, (size_t)D_ * D_);
  cvt(Wk, Wk16, (size_t)D_ * D_);
  cvt(Wv, Wv16, (size_t)D_ * D_);
  cvt(Wo, Wo16, (size_t)D_ * D_);
  cvt(W1, W116, (size_t)D_ * RANK_);
  cvt(W2, W216, (size_t)RANK_ * HID_);
  cvt(W3, W316, (size_t)HID_ * RANK_);
  cvt(W4, W416, (size_t)RANK_ * D_);

  // attention block
  ln_kernel<<<BS_, 256, 0, stream>>>(x, ln1g, ln1b, h1);
  dim3 gDD(D_ / BN, BS_ / BM);  // (16, 64)
  gemm_bf16_k<EPI_ELU1><<<gDD, 256, 0, stream>>>(h1, Wq16, bq, nullptr, nullptr, Qb, BS_, D_, D_);
  gemm_bf16_k<EPI_ELU1><<<gDD, 256, 0, stream>>>(h1, Wk16, bk, nullptr, nullptr, Kb, BS_, D_, D_);
  gemm_bf16_k<EPI_NONE><<<gDD, 256, 0, stream>>>(h1, Wv16, bv, nullptr, nullptr, Vb, BS_, D_, D_);
  kv_kernel<<<B_ * H_, 256, 0, stream>>>(Kb, Vb, kv16);
  colsum_kernel<<<B_ * H_, 128, 0, stream>>>(Kb, ksum);
  attn_kernel<<<dim3(B_ * H_, S_ / BM), 256, 0, stream>>>(Qb, kv16, ksum, attb);
  gemm_bf16_k<EPI_RESID><<<gDD, 256, 0, stream>>>(attb, Wo16, bo, x, x2, nullptr, BS_, D_, D_);

  // low-rank FFN block
  ln_kernel<<<BS_, 256, 0, stream>>>(x2, ln2g, ln2b, h2);
  dim3 gR(RANK_ / BN, BS_ / BM);  // (2, 64)
  dim3 gH(HID_ / BN, BS_ / BM);   // (64, 64)
  gemm_bf16_k<EPI_NONE><<<gR, 256, 0, stream>>>(h2, W116, b1, nullptr, nullptr, f1, BS_, RANK_, D_);
  gemm_bf16_k<EPI_GELU><<<gH, 256, 0, stream>>>(f1, W216, b2, nullptr, nullptr, f2, BS_, HID_, RANK_);
  gemm_bf16_k<EPI_NONE><<<gR, 256, 0, stream>>>(f2, W316, b3, nullptr, nullptr, f3, BS_, RANK_, HID_);
  gemm_bf16_k<EPI_RESID><<<gDD, 256, 0, stream>>>(f3, W416, b4, x2, out, nullptr, BS_, D_, RANK_);
}
